// GATNet_3582002725455
// MI455X (gfx1250) — matrix operand order repaired and checked by reading, not compiled, not run on hardware
//
#include <hip/hip_runtime.h>
#include <hip/hip_bf16.h>

// ---------------------------------------------------------------------------
// GATv2 + MLP + CNN pipeline for MI455X (gfx1250, wave32).
// Big GEMMs (16384x1600x1600 etc.) run on v_wmma_f32_16x16x32_f16 with f32
// accumulation. GEMM staging uses CDNA5 async global->LDS copies (ASYNCcnt)
// when the builtins are available, else a register double-buffer pipeline.
// ---------------------------------------------------------------------------

typedef _Float16 half_t;
typedef __attribute__((ext_vector_type(16))) _Float16 v16h;
typedef __attribute__((ext_vector_type(8)))  float    v8f;

#if __has_builtin(__builtin_amdgcn_global_load_async_to_lds_b128) && \
    __has_builtin(__builtin_amdgcn_s_wait_asynccnt)
#define USE_ASYNC 1
// builtin prototype (from hipcc diagnostic): pointee is a GCC-style int vector
// of 4 (vector_size(16)); global side is AS1, LDS side is AS3.
typedef int i4vs __attribute__((vector_size(16)));
typedef __attribute__((address_space(1))) i4vs gas_i4;
typedef __attribute__((address_space(3))) i4vs las_i4;
#else
#define USE_ASYNC 0
#endif

static constexpr int NNODE = 16384;
static constexpr int NEDGE = 65536;
static constexpr int NEL   = NEDGE + NNODE;   // edges + self loops = 81920
static constexpr int HC    = 1600;
static constexpr int NH    = 8;
static constexpr int CPH   = 200;
static constexpr int CNN   = 64;

// ---- workspace layout (bytes) ----
static constexpr size_t SZ_NODE = (size_t)NNODE * HC * 4;     // 104,857,600
static constexpr size_t OFF_XL   = 0;                          // xl ; later A1f16/A2f16
static constexpr size_t OFF_XR   = SZ_NODE;                    // xr ; later h2f16 (at +32MB)
static constexpr size_t OFF_H    = 2 * SZ_NODE;                // gat out ; later gemm1 out / conv bufs
static constexpr size_t R3       = 3 * SZ_NODE;
static constexpr size_t OFF_ALPHA= R3;                         // 81920*8*4
static constexpr size_t OFF_EX   = R3 + (4u<<20);
static constexpr size_t OFF_AMAX = R3 + (8u<<20);              // uint keys 131072
static constexpr size_t OFF_DEN  = R3 + (9u<<20);              // float 131072
static constexpr size_t OFF_PSUM = R3 + (10u<<20);             // 64*1600*4 max
static constexpr size_t OFF_PSQ  = R3 + (10u<<20) + (512u<<10);
static constexpr size_t OFF_SCL  = R3 + (11u<<20);             // 1600 f
static constexpr size_t OFF_SHF  = OFF_SCL + 8192;
static constexpr size_t OFF_EAP  = OFF_SHF + 8192;             // 256 partials
static constexpr size_t OFF_EAM  = OFF_EAP + 4096;             // mean scalar
static constexpr size_t OFF_W1H  = R3 + (12u<<20);             // 5,120,000
static constexpr size_t OFF_W2H  = R3 + (18u<<20);
static constexpr size_t OFF_WPH  = R3 + (24u<<20);             // 204,800
static constexpr size_t OFF_A1H  = OFF_XL;                     // 52 MB f16
static constexpr size_t OFF_H2H  = OFF_XR + (32u<<20);         // 52 MB f16
static constexpr size_t OFF_P    = OFF_H;                      // 4 MB  (gemm3 out, NHWC)
static constexpr size_t OFF_C1   = OFF_H + (8u<<20);           // 4 MB
static constexpr size_t OFF_C2   = OFF_H + (16u<<20);          // 8 MB
static constexpr size_t OFF_C3   = OFF_H + (32u<<20);          // 4 MB
static constexpr size_t WS_NEED  = OFF_WPH + 262144;

// ---------------------------------------------------------------------------
// small helpers
// ---------------------------------------------------------------------------
__global__ __launch_bounds__(256) void k_zero_u32(unsigned* __restrict__ p, int n) {
    int i = blockIdx.x * 256 + threadIdx.x;
    if (i < n) p[i] = 0u;
}

__global__ __launch_bounds__(256) void k_f32_to_f16(const float* __restrict__ s,
                                                    half_t* __restrict__ d, int n) {
    int i = blockIdx.x * 256 + threadIdx.x;
    if (i < n) d[i] = (half_t)s[i];
}

// monotonic float<->uint key for order-independent atomic max
__device__ __forceinline__ unsigned fkey(float f) {
    unsigned b = __float_as_uint(f);
    return (b & 0x80000000u) ? ~b : (b | 0x80000000u);
}
__device__ __forceinline__ float funkey(unsigned u) {
    return (u & 0x80000000u) ? __uint_as_float(u & 0x7fffffffu) : __uint_as_float(~u);
}

// ---------------------------------------------------------------------------
// GATv2 stage
// ---------------------------------------------------------------------------
__global__ __launch_bounds__(256) void k_gat_xlxr(
    const float* __restrict__ x, const float* __restrict__ Wl, const float* __restrict__ bl,
    const float* __restrict__ Wr, const float* __restrict__ br,
    float* __restrict__ xl, float* __restrict__ xr) {
    int n = blockIdx.x;
    __shared__ float xs[8];
    if (threadIdx.x < 7) xs[threadIdx.x] = x[n * 7 + threadIdx.x];
    __syncthreads();
    size_t base = (size_t)n * HC;
    for (int j = threadIdx.x; j < HC; j += 256) {
        float sl = bl[j], sr = br[j];
        #pragma unroll
        for (int i = 0; i < 7; ++i) {
            sl += xs[i] * Wl[i * HC + j];
            sr += xs[i] * Wr[i * HC + j];
        }
        xl[base + j] = sl;
        xr[base + j] = sr;
    }
}

__global__ __launch_bounds__(256) void k_ea_partial(const float* __restrict__ ea,
                                                    float* __restrict__ part) {
    __shared__ float s[256];
    int i = blockIdx.x * 256 + threadIdx.x;
    s[threadIdx.x] = ea[i];
    __syncthreads();
    for (int o = 128; o; o >>= 1) {
        if (threadIdx.x < o) s[threadIdx.x] += s[threadIdx.x + o];
        __syncthreads();
    }
    if (threadIdx.x == 0) part[blockIdx.x] = s[0];
}

__global__ void k_ea_final(const float* __restrict__ part, float* __restrict__ mean) {
    float s = 0.f;
    for (int i = 0; i < 256; ++i) s += part[i];
    *mean = s / (float)NEDGE;
}

// one block per edge; wave w handles head w (deterministic shuffle reduction)
__global__ __launch_bounds__(256) void k_gat_alpha(
    const float* __restrict__ xl, const float* __restrict__ xr,
    const float* __restrict__ ea, const float* __restrict__ We,
    const float* __restrict__ att, const int* __restrict__ src,
    const int* __restrict__ dst, const float* __restrict__ eamean,
    float* __restrict__ alpha) {
    int e = blockIdx.x;
    int s, d; float a;
    if (e < NEDGE) { s = src[e]; d = dst[e]; a = ea[e]; }
    else           { s = d = e - NEDGE;      a = *eamean; }
    int wv = threadIdx.x >> 5, lane = threadIdx.x & 31;
    const float* pl = xl + (size_t)s * HC + wv * CPH;
    const float* pr = xr + (size_t)d * HC + wv * CPH;
    const float* pw = We + wv * CPH;
    const float* pa = att + wv * CPH;
    float sum = 0.f;
    for (int c = lane; c < CPH; c += 32) {
        float m = pl[c] + pr[c] + a * pw[c];
        m = (m > 0.f) ? m : 0.2f * m;          // leaky_relu 0.2
        sum += m * pa[c];
    }
    for (int off = 16; off; off >>= 1) sum += __shfl_down(sum, off, 32);
    if (lane == 0) alpha[(size_t)e * NH + wv] = sum;
}

__global__ __launch_bounds__(256) void k_gat_amax(
    const float* __restrict__ alpha, const int* __restrict__ dst,
    unsigned* __restrict__ amaxkey) {
    int i = blockIdx.x * 256 + threadIdx.x;
    if (i >= NEL * NH) return;
    int e = i >> 3, h = i & 7;
    int d = (e < NEDGE) ? dst[e] : (e - NEDGE);
    atomicMax(&amaxkey[d * NH + h], fkey(alpha[i]));
}

__global__ __launch_bounds__(256) void k_gat_exdenom(
    const float* __restrict__ alpha, const int* __restrict__ dst,
    const unsigned* __restrict__ amaxkey, float* __restrict__ ex,
    float* __restrict__ denom) {
    int i = blockIdx.x * 256 + threadIdx.x;
    if (i >= NEL * NH) return;
    int e = i >> 3, h = i & 7;
    int d = (e < NEDGE) ? dst[e] : (e - NEDGE);
    float v = expf(alpha[i] - funkey(amaxkey[d * NH + h]));
    ex[i] = v;
    atomicAdd(&denom[d * NH + h], v);
}

__global__ __launch_bounds__(256) void k_gat_initout(const float* __restrict__ gat_b,
                                                     float* __restrict__ hout) {
    int i = blockIdx.x * 256 + threadIdx.x;   // N*HC total, exact grid
    hout[i] = gat_b[i % HC];
}

__global__ __launch_bounds__(256) void k_gat_agg(
    const float* __restrict__ xl, const int* __restrict__ src,
    const int* __restrict__ dst, const float* __restrict__ ex,
    const float* __restrict__ denom, float* __restrict__ hout) {
    int e = blockIdx.x;
    int s, d;
    if (e < NEDGE) { s = src[e]; d = dst[e]; }
    else           { s = d = e - NEDGE; }
    __shared__ float aw[NH];
    if (threadIdx.x < NH)
        aw[threadIdx.x] = ex[(size_t)e * NH + threadIdx.x] / denom[d * NH + threadIdx.x];
    __syncthreads();
    const float* pl = xl + (size_t)s * HC;
    float* po = hout + (size_t)d * HC;
    for (int j = threadIdx.x; j < HC; j += 256)
        atomicAdd(&po[j], pl[j] * aw[j / CPH]);
}

// ---------------------------------------------------------------------------
// BatchNorm (deterministic two-stage column stats) + fused applies
// ---------------------------------------------------------------------------
__global__ __launch_bounds__(256) void k_colsum_partial(
    const float* __restrict__ x, int Nrows, int C,
    float* __restrict__ psum, float* __restrict__ psq) {
    int c = blockIdx.x * 256 + threadIdx.x;
    if (c >= C) return;
    int chunk = Nrows / gridDim.y;
    int r0 = blockIdx.y * chunk;
    float s = 0.f, q = 0.f;
    for (int r = r0; r < r0 + chunk; ++r) {
        float v = x[(size_t)r * C + c];
        s += v; q += v * v;
    }
    psum[blockIdx.y * C + c] = s;
    psq [blockIdx.y * C + c] = q;
}

__global__ __launch_bounds__(256) void k_bn_finalize(
    const float* __restrict__ psum, const float* __restrict__ psq,
    const float* __restrict__ g, const float* __restrict__ b,
    int C, int P, float invN, float* __restrict__ scale, float* __restrict__ shift) {
    int c = blockIdx.x * 256 + threadIdx.x;
    if (c >= C) return;
    float s = 0.f, q = 0.f;
    for (int p = 0; p < P; ++p) { s += psum[p * C + c]; q += psq[p * C + c]; }
    float mu = s * invN;
    float var = q * invN - mu * mu;
    float sc = g[c] * rsqrtf(var + 1e-5f);
    scale[c] = sc;
    shift[c] = b[c] - mu * sc;
}

// bn + leaky_relu(0.01) + f16 convert (GEMM1 A prep)
__global__ __launch_bounds__(256) void k_prep_lrelu_f16(
    const float* __restrict__ x, const float* __restrict__ scale,
    const float* __restrict__ shift, half_t* __restrict__ d, int C) {
    int i = blockIdx.x * 256 + threadIdx.x;
    int c = i % C;
    float v = x[i] * scale[c] + shift[c];
    v = (v > 0.f) ? v : 0.01f * v;
    d[i] = (half_t)v;
}

// bn + relu + f16 convert (GEMM2 A prep)
__global__ __launch_bounds__(256) void k_prep_relu_f16(
    const float* __restrict__ x, const float* __restrict__ scale,
    const float* __restrict__ shift, half_t* __restrict__ d, int C) {
    int i = blockIdx.x * 256 + threadIdx.x;
    int c = i % C;
    float v = x[i] * scale[c] + shift[c];
    d[i] = (half_t)(v > 0.f ? v : 0.f);
}

// in-place bn + relu (conv stages, NHWC)
__global__ __launch_bounds__(256) void k_bnrelu_inplace(
    float* __restrict__ x, const float* __restrict__ scale,
    const float* __restrict__ shift, int C) {
    int i = blockIdx.x * 256 + threadIdx.x;
    int c = i % C;
    float v = x[i] * scale[c] + shift[c];
    x[i] = (v > 0.f) ? v : 0.f;
}

// ---------------------------------------------------------------------------
// WMMA GEMM: C[M,N] = A[M,K] * B[K,N] (+bias, optional relu+f16 store)
// block tile 128x64, 8 waves, each wave 2x2 fragments of 16x16x32 f16 WMMA.
// Staging: ping-pong LDS + GLOBAL_LOAD_ASYNC_TO_LDS_B128 (ASYNCcnt) when
// available, else VGPR double-buffer prefetch.
// MODE 0: store f32 (+bias). MODE 1: +bias, relu, store f16.
// ---------------------------------------------------------------------------
template <int MODE>
__global__ __launch_bounds__(256) void k_gemm_wmma(
    const half_t* __restrict__ A, const half_t* __restrict__ Bm,
    const float* __restrict__ bias, float* __restrict__ Cf32,
    half_t* __restrict__ Cf16, int M, int Nn, int K) {
    constexpr int BM = 128, BN = 64, BK = 32;
#if USE_ASYNC
    constexpr int NBUF = 2;
#else
    constexpr int NBUF = 1;
#endif
    __shared__ __align__(16) half_t As[NBUF][BM][BK + 8];
    __shared__ __align__(16) half_t Bs[NBUF][BK][BN + 8];

    const int tid = threadIdx.x;
    const int wv = tid >> 5, lane = tid & 31;
    const int bm = blockIdx.y * BM;
    const int bn = blockIdx.x * BN;
    const int wm = (wv & 3) * 32;   // wave offset in M
    const int wn = (wv >> 2) * 32;  // wave offset in N
    const int lh = lane >> 4;       // lane half
    const int lr = lane & 15;

    // staging coordinates: 2 A chunks (rows r, r+64) + 1 B chunk per thread
    const int ar = (tid * 8) >> 5;      // 0..63
    const int ac = (tid * 8) & 31;      // 0,8,16,24
    const int brr = tid >> 3;           // 0..31
    const int bcc = (tid & 7) * 8;      // 0..56

    v8f acc[2][2] = {};
    union F16x16 { v16h v; float4 f[2]; };
    const int nstep = K / BK;

#if USE_ASYNC
    auto issue = [&](int k0, int buf) {
        __builtin_amdgcn_global_load_async_to_lds_b128(
            (gas_i4*)(A + (size_t)(bm + ar) * K + k0 + ac),
            (las_i4*)&As[buf][ar][ac], 0, 0);
        __builtin_amdgcn_global_load_async_to_lds_b128(
            (gas_i4*)(A + (size_t)(bm + ar + 64) * K + k0 + ac),
            (las_i4*)&As[buf][ar + 64][ac], 0, 0);
        __builtin_amdgcn_global_load_async_to_lds_b128(
            (gas_i4*)(Bm + (size_t)(k0 + brr) * Nn + bn + bcc),
            (las_i4*)&Bs[buf][brr][bcc], 0, 0);
    };
    issue(0, 0);
#else
    float4 ra0, ra1, rb;
    auto gload = [&](int k0) {
        ra0 = *(const float4*)(A + (size_t)(bm + ar) * K + k0 + ac);
        ra1 = *(const float4*)(A + (size_t)(bm + ar + 64) * K + k0 + ac);
        rb  = *(const float4*)(Bm + (size_t)(k0 + brr) * Nn + bn + bcc);
    };
    auto lstore = [&]() {
        *(float4*)(&As[0][ar][ac])      = ra0;
        *(float4*)(&As[0][ar + 64][ac]) = ra1;
        *(float4*)(&Bs[0][brr][bcc])    = rb;
    };
    gload(0);
    lstore();
    __syncthreads();
#endif

    for (int s = 0; s < nstep; ++s) {
        const bool has_next = (s + 1 < nstep);
#if USE_ASYNC
        const int buf = s & 1;
        if (has_next) {
            issue((s + 1) * BK, buf ^ 1);
            __builtin_amdgcn_s_wait_asynccnt(3);  // current tile landed, next in flight
        } else {
            __builtin_amdgcn_s_wait_asynccnt(0);
        }
        __syncthreads();
#else
        const int buf = 0;
        if (has_next) gload((s + 1) * BK);        // prefetch into VGPRs during compute
#endif
        // A fragments: lane<16 -> K {0..7,16..23}; lane>=16 -> K {8..15,24..31}
        F16x16 af[2], bf[2];
        #pragma unroll
        for (int i = 0; i < 2; ++i) {
            int r = wm + i * 16 + lr;
            int kb = lh * 8;
            af[i].f[0] = *(const float4*)(&As[buf][r][kb]);
            af[i].f[1] = *(const float4*)(&As[buf][r][16 + kb]);
        }
        // B fragments: lane holds column N=lane%16; K {0..7,16..23} (lane<16) / {8..15,24..31}
        #pragma unroll
        for (int j = 0; j < 2; ++j) {
            int c0 = wn + j * 16 + lr;
            int kb = lh * 8;
            #pragma unroll
            for (int s2 = 0; s2 < 8; ++s2) {
                bf[j].v[s2]     = Bs[buf][kb + s2][c0];
                bf[j].v[8 + s2] = Bs[buf][16 + kb + s2][c0];
            }
        }
        #pragma unroll
        for (int i = 0; i < 2; ++i)
            #pragma unroll
            for (int j = 0; j < 2; ++j)
                acc[i][j] = __builtin_amdgcn_wmma_f32_16x16x32_f16(
                    false, af[i].v, false, bf[j].v, (short)0, acc[i][j], false, false);
        __syncthreads();
#if !USE_ASYNC
        if (has_next) {
            lstore();
            __syncthreads();
        }
#endif
    }

    // epilogue: C/D layout -> vgpr r: lanes 0-15 M=r, lanes 16-31 M=r+8; N=lane%16
    #pragma unroll
    for (int i = 0; i < 2; ++i) {
        #pragma unroll
        for (int j = 0; j < 2; ++j) {
            #pragma unroll
            for (int r = 0; r < 8; ++r) {
                int row = bm + wm + i * 16 + (lh ? r + 8 : r);
                int col = bn + wn + j * 16 + lr;
                float v = acc[i][j][r] + bias[col];
                if (MODE == 0) {
                    Cf32[(size_t)row * Nn + col] = v;
                } else {
                    Cf16[(size_t)row * Nn + col] = (half_t)(v > 0.f ? v : 0.f);
                }
            }
        }
    }
}

// ---------------------------------------------------------------------------
// direct 5x5 convs, NHWC activations, OIHW weights, pad=2, 32x32 images, B=16
// ---------------------------------------------------------------------------
__global__ __launch_bounds__(256) void k_conv5_nhwc(
    const float* __restrict__ in, const float* __restrict__ w,
    const float* __restrict__ bias, float* __restrict__ out, int Cin, int Cout) {
    int idx = blockIdx.x * 256 + threadIdx.x;
    int co = idx % Cout;
    int px = (idx / Cout) & 31;
    int py = (idx / (Cout * 32)) & 31;
    int b  = idx / (Cout * 1024);
    float acc = bias[co];
    for (int ky = 0; ky < 5; ++ky) {
        int yy = py + ky - 2;
        if ((unsigned)yy >= 32u) continue;
        for (int kx = 0; kx < 5; ++kx) {
            int xx = px + kx - 2;
            if ((unsigned)xx >= 32u) continue;
            const float* ip = in + (size_t)((b * 32 + yy) * 32 + xx) * Cin;
            const float* wp = w + (size_t)(co * Cin) * 25 + ky * 5 + kx;
            for (int ci = 0; ci < Cin; ++ci) acc += ip[ci] * wp[ci * 25];
        }
    }
    out[idx] = acc;
}

__global__ __launch_bounds__(256) void k_conv5_nhwc_to_nchw(
    const float* __restrict__ in, const float* __restrict__ w,
    const float* __restrict__ bias, float* __restrict__ out, int Cin, int Cout) {
    int idx = blockIdx.x * 256 + threadIdx.x;
    int co = idx % Cout;
    int px = (idx / Cout) & 31;
    int py = (idx / (Cout * 32)) & 31;
    int b  = idx / (Cout * 1024);
    float acc = bias[co];
    for (int ky = 0; ky < 5; ++ky) {
        int yy = py + ky - 2;
        if ((unsigned)yy >= 32u) continue;
        for (int kx = 0; kx < 5; ++kx) {
            int xx = px + kx - 2;
            if ((unsigned)xx >= 32u) continue;
            const float* ip = in + (size_t)((b * 32 + yy) * 32 + xx) * Cin;
            const float* wp = w + (size_t)(co * Cin) * 25 + ky * 5 + kx;
            for (int ci = 0; ci < Cin; ++ci) acc += ip[ci] * wp[ci * 25];
        }
    }
    out[(size_t)((b * Cout + co) * 32 + py) * 32 + px] = acc;
}

// ---------------------------------------------------------------------------
extern "C" void kernel_launch(void* const* d_in, const int* in_sizes, int n_in,
                              void* d_out, int out_size, void* d_ws, size_t ws_size,
                              hipStream_t stream) {
    (void)in_sizes; (void)n_in; (void)out_size;
    const float* x     = (const float*)d_in[0];
    const float* eattr = (const float*)d_in[1];
    const float* Wl    = (const float*)d_in[2];
    const float* bl    = (const float*)d_in[3];
    const float* Wr    = (const float*)d_in[4];
    const float* br    = (const float*)d_in[5];
    const float* We    = (const float*)d_in[6];
    const float* att   = (const float*)d_in[7];
    const float* gat_b = (const float*)d_in[8];
    const float* bn1_g = (const float*)d_in[9];
    const float* bn1_b = (const float*)d_in[10];
    const float* W1    = (const float*)d_in[11];
    const float* b1    = (const float*)d_in[12];
    const float* bn2_g = (const float*)d_in[13];
    const float* bn2_b = (const float*)d_in[14];
    const float* W2    = (const float*)d_in[15];
    const float* b2    = (const float*)d_in[16];
    const float* Wp    = (const float*)d_in[17];
    const float* bp    = (const float*)d_in[18];
    const float* cw1   = (const float*)d_in[19];
    const float* cb1   = (const float*)d_in[20];
    const float* g1    = (const float*)d_in[21];
    const float* be1   = (const float*)d_in[22];
    const float* cw2   = (const float*)d_in[23];
    const float* cb2   = (const float*)d_in[24];
    const float* g2    = (const float*)d_in[25];
    const float* be2   = (const float*)d_in[26];
    const float* cw3   = (const float*)d_in[27];
    const float* cb3   = (const float*)d_in[28];
    const float* g3    = (const float*)d_in[29];
    const float* be3   = (const float*)d_in[30];
    const float* cw4   = (const float*)d_in[31];
    const float* cb4   = (const float*)d_in[32];
    const int*   srcp  = (const int*)d_in[33];
    const int*   dstp  = (const int*)d_in[34];
    float* out = (float*)d_out;
    char* ws = (char*)d_ws;
    if (ws_size < WS_NEED) return;   // needs ~340 MB scratch

    float*    xl     = (float*)(ws + OFF_XL);
    float*    xr     = (float*)(ws + OFF_XR);
    float*    hbuf   = (float*)(ws + OFF_H);        // gat out, then gemm1 out
    float*    alpha  = (float*)(ws + OFF_ALPHA);
    float*    exb    = (float*)(ws + OFF_EX);
    unsigned* amaxk  = (unsigned*)(ws + OFF_AMAX);
    float*    denom  = (float*)(ws + OFF_DEN);
    float*    psum   = (float*)(ws + OFF_PSUM);
    float*    psq    = (float*)(ws + OFF_PSQ);
    float*    scl    = (float*)(ws + OFF_SCL);
    float*    shf    = (float*)(ws + OFF_SHF);
    float*    eapart = (float*)(ws + OFF_EAP);
    float*    eamean = (float*)(ws + OFF_EAM);
    half_t*   W1h    = (half_t*)(ws + OFF_W1H);
    half_t*   W2h    = (half_t*)(ws + OFF_W2H);
    half_t*   Wph    = (half_t*)(ws + OFF_WPH);
    half_t*   A1h    = (half_t*)(ws + OFF_A1H);     // also A2 (reused)
    half_t*   h2h    = (half_t*)(ws + OFF_H2H);
    float*    pbuf   = (float*)(ws + OFF_P);
    float*    c1     = (float*)(ws + OFF_C1);
    float*    c2     = (float*)(ws + OFF_C2);
    float*    c3     = (float*)(ws + OFF_C3);

    const int NHCblk = NNODE * HC / 256;            // 102400

    // ---- GATv2 ----
    k_zero_u32<<<512, 256, 0, stream>>>(amaxk, NNODE * NH);
    k_zero_u32<<<512, 256, 0, stream>>>((unsigned*)denom, NNODE * NH);
    k_gat_xlxr<<<NNODE, 256, 0, stream>>>(x, Wl, bl, Wr, br, xl, xr);
    k_ea_partial<<<256, 256, 0, stream>>>(eattr, eapart);
    k_ea_final<<<1, 1, 0, stream>>>(eapart, eamean);
    k_gat_alpha<<<NEL, 256, 0, stream>>>(xl, xr, eattr, We, att, srcp, dstp, eamean, alpha);
    k_gat_amax<<<(NEL * NH + 255) / 256, 256, 0, stream>>>(alpha, dstp, amaxk);
    k_gat_exdenom<<<(NEL * NH + 255) / 256, 256, 0, stream>>>(alpha, dstp, amaxk, exb, denom);
    k_gat_initout<<<NHCblk, 256, 0, stream>>>(gat_b, hbuf);
    k_gat_agg<<<NEL, 256, 0, stream>>>(xl, srcp, dstp, exb, denom, hbuf);

    // weight conversions (xl/xr dead after aggregation; W*h live in R3)
    k_f32_to_f16<<<HC * HC / 256, 256, 0, stream>>>(W1, W1h, HC * HC);
    k_f32_to_f16<<<HC * HC / 256, 256, 0, stream>>>(W2, W2h, HC * HC);
    k_f32_to_f16<<<HC * CNN / 256, 256, 0, stream>>>(Wp, Wph, HC * CNN);

    // ---- bn1 + leaky(0.01) -> f16 A1 ----
    k_colsum_partial<<<dim3(7, 64), 256, 0, stream>>>(hbuf, NNODE, HC, psum, psq);
    k_bn_finalize<<<7, 256, 0, stream>>>(psum, psq, bn1_g, bn1_b, HC, 64, 1.f / NNODE, scl, shf);
    k_prep_lrelu_f16<<<NHCblk, 256, 0, stream>>>(hbuf, scl, shf, A1h, HC);

    // ---- GEMM1: A1 @ W1 + b1 -> f32 (hbuf) ----
    k_gemm_wmma<0><<<dim3(HC / 64, NNODE / 128), 256, 0, stream>>>(
        A1h, W1h, b1, hbuf, (half_t*)nullptr, NNODE, HC, HC);

    // ---- bn2 + relu -> f16 A2 ----
    k_colsum_partial<<<dim3(7, 64), 256, 0, stream>>>(hbuf, NNODE, HC, psum, psq);
    k_bn_finalize<<<7, 256, 0, stream>>>(psum, psq, bn2_g, bn2_b, HC, 64, 1.f / NNODE, scl, shf);
    k_prep_relu_f16<<<NHCblk, 256, 0, stream>>>(hbuf, scl, shf, A1h, HC);

    // ---- GEMM2: A2 @ W2 + b2, relu -> f16 (h2h) ----
    k_gemm_wmma<1><<<dim3(HC / 64, NNODE / 128), 256, 0, stream>>>(
        A1h, W2h, b2, (float*)nullptr, h2h, NNODE, HC, HC);

    // ---- GEMM3: h2 @ Wp + bp -> f32 NHWC [16,32,32,64] ----
    k_gemm_wmma<0><<<dim3(1, NNODE / 128), 256, 0, stream>>>(
        h2h, Wph, bp, pbuf, (half_t*)nullptr, NNODE, CNN, HC);

    // ---- conv stack (NHWC) ----
    k_conv5_nhwc<<<16 * 1024 * 64 / 256, 256, 0, stream>>>(pbuf, cw1, cb1, c1, 64, 64);
    k_colsum_partial<<<dim3(1, 64), 256, 0, stream>>>(c1, NNODE, 64, psum, psq);
    k_bn_finalize<<<1, 256, 0, stream>>>(psum, psq, g1, be1, 64, 64, 1.f / NNODE, scl, shf);
    k_bnrelu_inplace<<<NNODE * 64 / 256, 256, 0, stream>>>(c1, scl, shf, 64);

    k_conv5_nhwc<<<16 * 1024 * 128 / 256, 256, 0, stream>>>(c1, cw2, cb2, c2, 64, 128);
    k_colsum_partial<<<dim3(1, 64), 256, 0, stream>>>(c2, NNODE, 128, psum, psq);
    k_bn_finalize<<<1, 256, 0, stream>>>(psum, psq, g2, be2, 128, 64, 1.f / NNODE, scl, shf);
    k_bnrelu_inplace<<<NNODE * 128 / 256, 256, 0, stream>>>(c2, scl, shf, 128);

    k_conv5_nhwc<<<16 * 1024 * 64 / 256, 256, 0, stream>>>(c2, cw3, cb3, c3, 128, 64);
    k_colsum_partial<<<dim3(1, 64), 256, 0, stream>>>(c3, NNODE, 64, psum, psq);
    k_bn_finalize<<<1, 256, 0, stream>>>(psum, psq, g3, be3, 64, 64, 1.f / NNODE, scl, shf);
    k_bnrelu_inplace<<<NNODE * 64 / 256, 256, 0, stream>>>(c3, scl, shf, 64);

    k_conv5_nhwc_to_nchw<<<16 * 1024 * 6 / 256, 256, 0, stream>>>(c3, cw4, cb4, out, 64, 6);
}
